// NNGcnConv_16149077033572
// MI455X (gfx1250) — matrix operand order repaired and checked by reading, not compiled, not run on hardware
//
#include <hip/hip_runtime.h>
#include <hip/hip_bf16.h>
#include <math.h>

typedef __attribute__((ext_vector_type(16))) _Float16 v16h;
typedef __attribute__((ext_vector_type(8)))  float    v8f;

union HFrag { v16h v; _Float16 h[16]; };
union CFrag { v8f  v; float    f[8];  };

// ---------------------------------------------------------------------------
// GEMM: C[M,Nw] = A[M,K] @ W[K,Nw]  (+bias, relu/tanh epilogue variants)
// K, Nw compile-time so the K-loop is branch-free and fully unrolled.
// One wave computes one 16x16 tile via v_wmma_f32_16x16x32_f16.
// W is staged in LDS transposed [col][k], zero-padded to [NP][KP], so the
// B fragment is two 16B LDS reads per lane (k0+ka.., k0+16+ka..) (ISA B layout).
// EPI: 0 = plain store, 1 = +bias,relu, 2 = +bias,tanh
// ---------------------------------------------------------------------------
template<int K, int Nw, int EPI>
__global__ __launch_bounds__(128)
void gemm_wmma_kernel(const float* __restrict__ A, const float* __restrict__ W,
                      const float* __restrict__ bias, float* __restrict__ C, int M)
{
    constexpr int KP = (K + 31) & ~31;     // K padded to 32
    constexpr int NP = (Nw + 15) & ~15;    // Nw padded to 16
    __shared__ __align__(32) _Float16 Wl[NP * KP];   // transposed, zero-padded

    // Cooperative fill (handles Nw/K padding once, outside the hot loop)
    for (int i = threadIdx.x; i < NP * KP; i += 128) {
        const int cc = i / KP, kk = i % KP;
        Wl[i] = (kk < K && cc < Nw) ? (_Float16)W[(size_t)kk * Nw + cc]
                                    : (_Float16)0.0f;
    }
    __syncthreads();

    const int lane = threadIdx.x & 31;
    const int wave = threadIdx.x >> 5;
    const int rowTile = (blockIdx.x * 4 + wave) * 16;
    if (rowTile >= M) return;              // wave-uniform: EXEC stays all-1s
    const int colTile = blockIdx.y * 16;

    const int r  = lane & 15;              // A: row within tile
    const int ka = (lane >> 4) * 8;        // A: k-group offset (0 or 8)
    const int kb = (lane >> 4) * 16;       // unused (0 or 16); B uses ka like A
    const int colP = colTile + (lane & 15);

    const float*    Arow = A + (size_t)(rowTile + r) * K;
    const _Float16* Bcol = &Wl[colP * KP];

    CFrag c = {};                          // f32 accumulator

#pragma unroll
    for (int k0 = 0; k0 < K; k0 += 32) {
        HFrag a, b;

        // A lower half: K = k0+ka .. k0+ka+7  (contiguous -> 2x float4)
        {
            float t[8];
            *(float4*)&t[0] = *(const float4*)&Arow[k0 + ka];
            *(float4*)&t[4] = *(const float4*)&Arow[k0 + ka + 4];
#pragma unroll
            for (int j = 0; j < 8; ++j) a.h[j] = (_Float16)t[j];
        }
        // A upper half: K = k0+16+ka .. +7 (folds away when K == 16)
        if (k0 + 16 < K) {
            float t[8];
            *(float4*)&t[0] = *(const float4*)&Arow[k0 + 16 + ka];
            *(float4*)&t[4] = *(const float4*)&Arow[k0 + 16 + ka + 4];
#pragma unroll
            for (int j = 0; j < 8; ++j) a.h[8 + j] = (_Float16)t[j];
        } else {
#pragma unroll
            for (int j = 0; j < 8; ++j) a.h[8 + j] = (_Float16)0.0f;
        }

        // B fragment: halves [colP][k0+ka..+7] and [colP][k0+16+ka..+7] (matches A)
#pragma unroll
        for (int j = 0; j < 8; ++j) { b.h[j] = Bcol[k0 + ka + j]; b.h[8 + j] = Bcol[k0 + 16 + ka + j]; }

        c.v = __builtin_amdgcn_wmma_f32_16x16x32_f16(
                  /*neg_a=*/false, a.v, /*neg_b=*/false, b.v,
                  /*c_mod=*/(short)0, c.v, /*reuse_a=*/false, /*reuse_b=*/false);
    }

    // D layout: VGPR rr -> row = rowTile + rr + (lane>=16 ? 8 : 0), col = colP
    if (colP < Nw) {
        const int rbase = rowTile + ((lane >> 4) * 8);
        const float bv = (EPI > 0) ? bias[colP] : 0.0f;
#pragma unroll
        for (int rr = 0; rr < 8; ++rr) {
            float v = c.f[rr];
            if (EPI == 1) { v += bv; v = v > 0.0f ? v : 0.0f; }
            if (EPI == 2) { v = tanhf(v + bv); }
            C[(size_t)(rbase + rr) * Nw + colP] = v;
        }
    }
}

// ---------------------------------------------------------------------------
// Degree accumulation: one atomic per edge (deg pre-zeroed).
// ---------------------------------------------------------------------------
__global__ void deg_kernel(const long long* __restrict__ dst,
                           float* __restrict__ deg, long long E)
{
    long long e = (long long)blockIdx.x * blockDim.x + threadIdx.x;
    if (e < E) atomicAdd(&deg[dst[e]], 1.0f);
}

__global__ void rsqrt_kernel(const float* __restrict__ deg,
                             float* __restrict__ dis, int N)
{
    int i = blockIdx.x * blockDim.x + threadIdx.x;
    if (i < N) dis[i] = rsqrtf(deg[i] + 1.0f);
}

// ---------------------------------------------------------------------------
// Edge aggregation: agg[dst, f] += h[src, f] * dis[src] * dis[dst].
// One thread per (edge, feature); F is a power of two (16/32/64/128).
// Gathers and scatter atomics resolve in the 192MB L2 (activations <= 51MB).
// ---------------------------------------------------------------------------
__global__ void edge_agg_kernel(const float* __restrict__ h,
                                const long long* __restrict__ src,
                                const long long* __restrict__ dst,
                                const float* __restrict__ dis,
                                float* __restrict__ agg,
                                long long E, int lgF)
{
    long long idx = (long long)blockIdx.x * blockDim.x + threadIdx.x;
    if (idx >= (E << lgF)) return;
    const int F = 1 << lgF;
    const long long e = idx >> lgF;
    const int f = (int)(idx & (long long)(F - 1));
    const long long s = src[e], d = dst[e];
    const float norm = dis[s] * dis[d];
    atomicAdd(&agg[d * F + f], h[s * F + f] * norm);
}

// ---------------------------------------------------------------------------
// Self-loop + bias + ReLU: agg = relu(agg + h * dis^2 + b)
// ---------------------------------------------------------------------------
__global__ void combine_kernel(float* __restrict__ agg, const float* __restrict__ h,
                               const float* __restrict__ dis, const float* __restrict__ b,
                               long long total, int lgF)
{
    long long idx = (long long)blockIdx.x * blockDim.x + threadIdx.x;
    if (idx >= total) return;
    const int F = 1 << lgF;
    const long long i = idx >> lgF;
    const int f = (int)(idx & (long long)(F - 1));
    const float s = dis[i];
    float v = agg[idx] + h[idx] * s * s + b[f];
    agg[idx] = v > 0.0f ? v : 0.0f;
}

// ---------------------------------------------------------------------------
extern "C" void kernel_launch(void* const* d_in, const int* in_sizes, int n_in,
                              void* d_out, int out_size, void* d_ws, size_t ws_size,
                              hipStream_t stream)
{
    const float*     x   = (const float*)d_in[0];
    const long long* ei  = (const long long*)d_in[1];
    const float* W1  = (const float*)d_in[2];   const float* b1  = (const float*)d_in[3];
    const float* W2  = (const float*)d_in[4];   const float* b2  = (const float*)d_in[5];
    const float* W3  = (const float*)d_in[6];   const float* b3  = (const float*)d_in[7];
    const float* W4  = (const float*)d_in[8];   const float* b4  = (const float*)d_in[9];
    const float* Wf1 = (const float*)d_in[10];  const float* bf1 = (const float*)d_in[11];
    const float* Wf2 = (const float*)d_in[12];  const float* bf2 = (const float*)d_in[13];

    const int N = in_sizes[0] / 128;            // 100000
    const long long E = in_sizes[1] / 2;        // 1600000
    const long long* src = ei;
    const long long* dst = ei + E;

    // Workspace layout (floats): dis[N], deg[N], hA[N*128], hB[N*128]
    float* dis = (float*)d_ws;
    float* deg = dis + N;
    float* hA  = deg + N;                       // dense GEMM output h
    float* hB  = hA + (size_t)N * 128;          // aggregation / activation buffer

    const int rowTiles = (N + 15) / 16;         // 6250
    const int gx = (rowTiles + 3) / 4;          // 4 waves (tiles) per block

    // ---- degrees -> deg_inv_sqrt ----
    hipMemsetAsync(deg, 0, (size_t)N * sizeof(float), stream);
    deg_kernel<<<(int)((E + 255) / 256), 256, 0, stream>>>(dst, deg, E);
    rsqrt_kernel<<<(N + 255) / 256, 256, 0, stream>>>(deg, dis, N);

    auto edge_phase = [&](const float* W, const float* b, int F, int lgF) {
        hipMemsetAsync(hB, 0, (size_t)N * F * sizeof(float), stream);
        const long long totE = E << lgF;
        edge_agg_kernel<<<(int)((totE + 255) / 256), 256, 0, stream>>>(
            hA, src, dst, dis, hB, E, lgF);
        const long long totN = (long long)N << lgF;
        combine_kernel<<<(int)((totN + 255) / 256), 256, 0, stream>>>(
            hB, hA, dis, b, totN, lgF);
    };

    // ---- layer 1: 128 -> 16 ----
    gemm_wmma_kernel<128, 16, 0><<<dim3(gx, 1), 128, 0, stream>>>(x,  W1, nullptr, hA, N);
    edge_phase(W1, b1, 16, 4);
    // ---- layer 2: 16 -> 32 ----
    gemm_wmma_kernel<16, 32, 0><<<dim3(gx, 2), 128, 0, stream>>>(hB, W2, nullptr, hA, N);
    edge_phase(W2, b2, 32, 5);
    // ---- layer 3: 32 -> 64 ----
    gemm_wmma_kernel<32, 64, 0><<<dim3(gx, 4), 128, 0, stream>>>(hB, W3, nullptr, hA, N);
    edge_phase(W3, b3, 64, 6);
    // ---- layer 4: 64 -> 128 ----
    gemm_wmma_kernel<64, 128, 0><<<dim3(gx, 8), 128, 0, stream>>>(hB, W4, nullptr, hA, N);
    edge_phase(W4, b4, 128, 7);

    // ---- dense head ----
    gemm_wmma_kernel<128, 32, 1><<<dim3(gx, 2), 128, 0, stream>>>(hB, Wf1, bf1, hA, N);
    gemm_wmma_kernel<32, 10, 2><<<dim3(gx, 1), 128, 0, stream>>>(hA, Wf2, bf2,
                                                                 (float*)d_out, N);
}
